// ParallelBERTGenerator_84104049590693
// MI455X (gfx1250) — compile-verified, matrix operand order checked by reading, not run on hardware
//
#include <hip/hip_runtime.h>
#include <math.h>

// MI455X / gfx1250: wave32, WMMA 16x16x32 bf16 with f32 accumulate.
// B=8, SQ=2048, SK=768, D=768.
//
// Simplification used (exact, not approximate):
//   out = concat(scores, scores@prev) @ Wc.T + bc
//       = scores @ (Wc[:, :D].T + prev @ Wc[:, D:].T) + bc
//       = scores @ M + bc
// Mask fold: masked k contributes exactly (-1e9)*M[k,:], independent of the
// score value, so we zero masked score entries and add
//   corr[b,i] = -1e9 * sum_{k: mask[b,k]==0} M[k,i]
// in the final epilogue. Keeps +-1e9 out of the bf16 WMMA datapath.

typedef __attribute__((ext_vector_type(8)))  float  v8f;
typedef __attribute__((ext_vector_type(16))) __bf16 v16bf;
typedef __attribute__((ext_vector_type(4)))  int    v4i;

#if defined(__has_builtin)
#if __has_builtin(__builtin_amdgcn_global_load_async_to_lds_b128)
#define HAVE_ASYNC_LDS 1
#endif
#if __has_builtin(__builtin_amdgcn_s_wait_asynccnt)
#define HAVE_WAIT_ASYNC 1
#endif
#endif

union FragBF { v16bf v; unsigned int u[8]; };

// Let the compiler pick v_cvt_pk_bf16_f32 (RNE) instead of manual bit math.
__device__ __forceinline__ unsigned int pack_bf16x2(float a, float b) {
  union { __bf16 h[2]; unsigned int u; } p;
  p.h[0] = (__bf16)a;
  p.h[1] = (__bf16)b;
  return p.u;
}
__device__ __forceinline__ unsigned short f32_to_bf16(float a) {
  union { __bf16 h; unsigned short s; } p;
  p.h = (__bf16)a;
  return p.s;
}

__device__ __forceinline__ void async_cp_32B(const void* gsrc, void* ldst) {
#if defined(HAVE_ASYNC_LDS)
  typedef __attribute__((address_space(1))) v4i* gp_t;
  typedef __attribute__((address_space(3))) v4i* lp_t;
  gp_t g = (gp_t)gsrc;
  lp_t l = (lp_t)ldst;
  __builtin_amdgcn_global_load_async_to_lds_b128(g, l, 0, 0);
  __builtin_amdgcn_global_load_async_to_lds_b128(g, l, 16, 0);
#else
  const unsigned int* s = (const unsigned int*)gsrc;
  unsigned int* d = (unsigned int*)ldst;
  #pragma unroll
  for (int j = 0; j < 8; ++j) d[j] = s[j];
#endif
}

__device__ __forceinline__ void wait_async0() {
#if defined(HAVE_WAIT_ASYNC)
  __builtin_amdgcn_s_wait_asynccnt(0);
#elif defined(HAVE_ASYNC_LDS)
  asm volatile("s_wait_asynccnt 0" ::: "memory");
#endif
}

// Tiled bf16 WMMA GEMM: C[m,n] = sum_k A[m,k] * B(k,n)
//   B_TRANS=1: B source is row-major [n, k] (weights / K^T case)
//   B_TRANS=0: B source is row-major [k, n]
// EPI: 0 = bf16( acc + bias[n] )
//      1 = mask[b,n]==0 ? 0 : bf16( acc * scale )
//      2 = f32 ( acc + bias[n] + corr[b,n] )
//      3 = v = acc + addT[n*ldAdd + m];  bf16 store + f32 store (M precompute)
// Block: 256 threads (8 waves). Tile 128x128, BK=32. Wave -> 32x64 patch
// (2x4 tiles of 16x16). All launched dims are multiples of the tile sizes.
template<int EPI, bool A_BF16, bool B_TRANS, bool B_BF16>
__launch_bounds__(256, 2)
__global__ void gemm_wmma_bf16(
    const void* __restrict__ Ap, long lda, long aBatch,
    const void* __restrict__ Bp, long ldb, long bBatch,
    void*       __restrict__ Op, long ldo, long oBatch,
    int Ktot,
    const float* __restrict__ bias,
    float scale,
    const int*   __restrict__ mask, long maskBatch,
    const float* __restrict__ corr, long corrBatch,
    const float* __restrict__ addT, long ldAdd,
    float* __restrict__ Ofp)
{
  // Row stride 17 dwords: 17 coprime with 64 banks -> conflict-free fragment
  // gathers (16 consecutive rows, same column).
  __shared__ unsigned int ldsA[128 * 17];
  __shared__ unsigned int ldsB[128 * 17];

  const int bz    = blockIdx.z;
  const int tileN = blockIdx.x * 128;
  const int tileM = blockIdx.y * 128;
  const int tid   = threadIdx.x;
  const int wave  = tid >> 5;
  const int lane  = tid & 31;
  const int lhalf = lane >> 4;   // 0: lanes 0-15, 1: lanes 16-31
  const int lmod  = lane & 15;
  const int wm    = (wave >> 1) * 32;   // wave row base in tile
  const int wn    = (wave & 1) * 64;    // wave col base in tile

  const float*          Af = (const float*)Ap          + (size_t)bz * aBatch;
  const unsigned short* Ah = (const unsigned short*)Ap + (size_t)bz * aBatch;
  const float*          Bf = (const float*)Bp          + (size_t)bz * bBatch;
  const unsigned short* Bh = (const unsigned short*)Bp + (size_t)bz * bBatch;

  const bool useAsync = (A_BF16 || (B_TRANS && B_BF16));

  v8f acc[2][4] = {};

  for (int k0 = 0; k0 < Ktot; k0 += 32) {
    __syncthreads();   // protect LDS from previous iteration's readers

    // ---- stage A tile: 128 rows x 32 k, packed as bf16 pairs along k ----
    {
      const int row  = tid >> 1;          // 0..127
      const int half = (tid & 1) * 16;    // k sub-offset
      unsigned int* dst = &ldsA[row * 17 + (half >> 1)];
      if (A_BF16) {
        const unsigned short* src = Ah + (size_t)(tileM + row) * lda + k0 + half;
        async_cp_32B(src, dst);           // pure byte copy: async DMA to LDS
        if (k0 + 32 < Ktot) __builtin_prefetch((const void*)(src + 32), 0, 1);
      } else {
        const float* src = Af + (size_t)(tileM + row) * lda + k0 + half;
        #pragma unroll
        for (int j = 0; j < 8; ++j) dst[j] = pack_bf16x2(src[2 * j], src[2 * j + 1]);
        if (k0 + 32 < Ktot) __builtin_prefetch((const void*)(src + 32), 0, 1);
      }
    }

    // ---- stage B tile: LdsB[n][j] = pack(B(2j,n), B(2j+1,n)) ----
    if (B_TRANS) {
      const int row  = tid >> 1;          // n index
      const int half = (tid & 1) * 16;
      unsigned int* dst = &ldsB[row * 17 + (half >> 1)];
      if (B_BF16) {
        const unsigned short* src = Bh + (size_t)(tileN + row) * ldb + k0 + half;
        async_cp_32B(src, dst);           // pure byte copy: async DMA to LDS
        if (k0 + 32 < Ktot) __builtin_prefetch((const void*)(src + 32), 0, 1);
      } else {
        const float* src = Bf + (size_t)(tileN + row) * ldb + k0 + half;
        #pragma unroll
        for (int j = 0; j < 8; ++j) dst[j] = pack_bf16x2(src[2 * j], src[2 * j + 1]);
        if (k0 + 32 < Ktot) __builtin_prefetch((const void*)(src + 32), 0, 1);
      }
    } else {
      // K-major source needs a transpose-pack; manual gather (B here is the
      // tiny 768x768 M matrix -> L2 resident).
      #pragma unroll
      for (int jj = 0; jj < 8; ++jj) {
        const int idx = tid + jj * 256;      // 2048 dwords total
        const int n   = idx & 127;
        const int j   = idx >> 7;            // k-pair index 0..15
        if (B_BF16) {
          unsigned int lo = Bh[(size_t)(k0 + 2 * j    ) * ldb + tileN + n];
          unsigned int hi = Bh[(size_t)(k0 + 2 * j + 1) * ldb + tileN + n];
          ldsB[n * 17 + j] = lo | (hi << 16);
        } else {
          float lo = Bf[(size_t)(k0 + 2 * j    ) * ldb + tileN + n];
          float hi = Bf[(size_t)(k0 + 2 * j + 1) * ldb + tileN + n];
          ldsB[n * 17 + j] = pack_bf16x2(lo, hi);
        }
      }
    }
    if (useAsync) wait_async0();   // ASYNCcnt==0: our LDS writes landed
    __syncthreads();

    // ---- fragments per ISA 16-bit A/B layout ----
    // lanes 0-15: K pairs {0..7} in u[0..3], {16..23} in u[4..7]
    // lanes 16-31: K pairs {8..15} / {24..31}
    FragBF fa[2], fb[4];
    #pragma unroll
    for (int mi = 0; mi < 2; ++mi) {
      const unsigned int* pr = &ldsA[(wm + mi * 16 + lmod) * 17];
      #pragma unroll
      for (int r = 0; r < 4; ++r) {
        fa[mi].u[r]     = pr[lhalf * 4 + r];
        fa[mi].u[r + 4] = pr[8 + lhalf * 4 + r];
      }
    }
    #pragma unroll
    for (int ni = 0; ni < 4; ++ni) {
      const unsigned int* pr = &ldsB[(wn + ni * 16 + lmod) * 17];
      #pragma unroll
      for (int r = 0; r < 4; ++r) {
        fb[ni].u[r]     = pr[lhalf * 4 + r];
        fb[ni].u[r + 4] = pr[8 + lhalf * 4 + r];
      }
    }

    #pragma unroll
    for (int mi = 0; mi < 2; ++mi)
      #pragma unroll
      for (int ni = 0; ni < 4; ++ni)
        acc[mi][ni] = __builtin_amdgcn_wmma_f32_16x16x32_bf16(
            false, fa[mi].v, false, fb[ni].v, (short)0, acc[mi][ni],
            false, false);
  }

  // ---- epilogue: C/D layout -> lane l, vgpr v: m = v + 8*(l>>4), n = l&15 ----
  unsigned short* Oh = (unsigned short*)Op + (size_t)bz * oBatch;
  float*          Of = (float*)Op          + (size_t)bz * oBatch;
  #pragma unroll
  for (int mi = 0; mi < 2; ++mi)
    #pragma unroll
    for (int ni = 0; ni < 4; ++ni)
      #pragma unroll
      for (int v = 0; v < 8; ++v) {
        const int m = tileM + wm + mi * 16 + lhalf * 8 + v;
        const int n = tileN + wn + ni * 16 + lmod;
        const float a = acc[mi][ni][v];
        if (EPI == 0) {
          Oh[(size_t)m * ldo + n] = f32_to_bf16(a + bias[n]);
        } else if (EPI == 1) {
          const int mv = mask[(size_t)bz * maskBatch + n];
          Oh[(size_t)m * ldo + n] = mv ? f32_to_bf16(a * scale) : (unsigned short)0;
        } else if (EPI == 2) {
          Of[(size_t)m * ldo + n] = a + bias[n] + corr[(size_t)bz * corrBatch + n];
        } else { // EPI == 3: M precompute, dual bf16 + f32 store
          const float vv = a + addT[(size_t)n * ldAdd + m];
          Oh [(size_t)m * ldo + n] = f32_to_bf16(vv);
          Ofp[(size_t)m * ldo + n] = vv;
        }
      }
}

// corr[b,n] = -1e9 * sum_{k: mask[b,k]==0} Mf[k,n]
__global__ void mask_corr_kernel(const float* __restrict__ Mf,
                                 const int*   __restrict__ mask,
                                 float*       __restrict__ corr,
                                 int SK, int D) {
  const int n = blockIdx.x * 256 + threadIdx.x;
  const int b = blockIdx.y;
  if (n >= D) return;
  float s = 0.0f;
  for (int k = 0; k < SK; ++k)
    if (mask[b * SK + k] == 0) s += Mf[(size_t)k * D + n];
  corr[(size_t)b * D + n] = -1.0e9f * s;
}

extern "C" void kernel_launch(void* const* d_in, const int* in_sizes, int n_in,
                              void* d_out, int out_size, void* d_ws, size_t ws_size,
                              hipStream_t stream) {
  (void)in_sizes; (void)n_in; (void)out_size; (void)ws_size;
  constexpr int B = 8, SQ = 2048, SK = 768, D = 768;

  const float* query = (const float*)d_in[0];
  const float* keym  = (const float*)d_in[1];
  // d_in[2] (value_matrix), d_in[9] (Wv), d_in[10] (bv): dead in the reference output
  const float* prev  = (const float*)d_in[3];
  const int*   amask = (const int*)d_in[4];
  const float* Wq = (const float*)d_in[5];
  const float* bq = (const float*)d_in[6];
  const float* Wk = (const float*)d_in[7];
  const float* bk = (const float*)d_in[8];
  const float* Wc = (const float*)d_in[11];
  const float* bc = (const float*)d_in[12];

  // Workspace layout (bytes); total ~63.3 MB
  char* ws = (char*)d_ws;
  unsigned short* Qp = (unsigned short*)(ws);              // [B*SQ, D] bf16  25,165,824 B
  unsigned short* Kp = (unsigned short*)(ws + 25165824);   // [B*SK, D] bf16   9,437,184 B
  unsigned short* Sc = (unsigned short*)(ws + 34603008);   // [B, SQ, SK] bf16 25,165,824 B
  unsigned short* Mb = (unsigned short*)(ws + 59768832);   // [SK, D] bf16     1,179,648 B
  float*          Mf = (float*)         (ws + 60948480);   // [SK, D] f32      2,359,296 B
  float*          Cr = (float*)         (ws + 63307776);   // [B, D]  f32         24,576 B

  const float scale = 1.0f / sqrtf((float)D);

  // 1) M[k,i] = Wc[i,k] + sum_d prev[k,d] * Wc[i, D+d]   (bf16 + f32 copies)
  gemm_wmma_bf16<3, false, true, false>
      <<<dim3(D / 128, SK / 128, 1), 256, 0, stream>>>(
      prev, D, 0, Wc + D, 2 * D, 0, Mb, D, 0, D,
      nullptr, 0.0f, nullptr, 0, nullptr, 0, Wc, 2 * D, Mf);

  // 2) mask correction from f32 M
  mask_corr_kernel<<<dim3(D / 256, B, 1), 256, 0, stream>>>(Mf, amask, Cr, SK, D);

  // 3) Q = query @ Wq.T + bq  (batch flattened into rows) -> bf16
  gemm_wmma_bf16<0, false, true, false>
      <<<dim3(D / 128, (B * SQ) / 128, 1), 256, 0, stream>>>(
      query, D, 0, Wq, D, 0, Qp, D, 0, D,
      bq, 0.0f, nullptr, 0, nullptr, 0, nullptr, 0, nullptr);

  // 4) K = key_matrix @ Wk.T + bk -> bf16
  gemm_wmma_bf16<0, false, true, false>
      <<<dim3(D / 128, (B * SK) / 128, 1), 256, 0, stream>>>(
      keym, D, 0, Wk, D, 0, Kp, D, 0, D,
      bk, 0.0f, nullptr, 0, nullptr, 0, nullptr, 0, nullptr);

  // 5) scores = (Q @ K^T) * scale, masked entries -> 0, bf16 (batched)
  gemm_wmma_bf16<1, true, true, true>
      <<<dim3(SK / 128, SQ / 128, B), 256, 0, stream>>>(
      Qp, D, (long)SQ * D, Kp, D, (long)SK * D, Sc, SK, (long)SQ * SK, D,
      nullptr, scale, amask, SK, nullptr, 0, nullptr, 0, nullptr);

  // 6) out = scores @ M + bc + corr  -> f32 d_out (batched)
  gemm_wmma_bf16<2, true, false, true>
      <<<dim3(D / 128, SQ / 128, B), 256, 0, stream>>>(
      Sc, SK, (long)SQ * SK, Mb, D, 0, d_out, D, (long)SQ * D, SK,
      bc, 0.0f, nullptr, 0, Cr, D, nullptr, 0, nullptr);
}
